// FlexCausalSelfAttention_25658134626323
// MI455X (gfx1250) — matrix operand order repaired and checked by reading, not compiled, not run on hardware
//
#include <hip/hip_runtime.h>

#define BB 2
#define SS 2048
#define EE 1024
#define HH 16
#define HD 64
#define PP 2048
#define MM (BB*SS)   // 4096

typedef __attribute__((ext_vector_type(16))) __bf16          v16bf;
typedef __attribute__((ext_vector_type(8)))  float           v8f;
typedef __attribute__((ext_vector_type(8)))  unsigned short  u16x8;
typedef __attribute__((ext_vector_type(16))) unsigned short  u16x16;

__device__ __forceinline__ unsigned short f2bf(float f) {
  unsigned u = __builtin_bit_cast(unsigned, f);
  u += 0x7FFFu + ((u >> 16) & 1u);           // round-to-nearest-even
  return (unsigned short)(u >> 16);
}

__device__ __forceinline__ v8f vzero() {
  v8f z;
#pragma unroll
  for (int i = 0; i < 8; ++i) z[i] = 0.0f;
  return z;
}

__device__ __forceinline__ v16bf make_frag(u16x8 lo, u16x8 hi) {
  u16x16 u;
#pragma unroll
  for (int i = 0; i < 8; ++i) { u[i] = lo[i]; u[i + 8] = hi[i]; }
  return __builtin_bit_cast(v16bf, u);
}

__device__ __forceinline__ v8f bwmma(v16bf a, v16bf b, v8f c) {
  return __builtin_amdgcn_wmma_f32_16x16x32_bf16(
      /*neg_a=*/false, a, /*neg_b=*/false, b,
      /*c_mod=*/(short)0, c, /*reuse_a=*/false, /*reuse_b=*/false);
}

// ---------------- prep kernels ----------------

__global__ void k_cvt_bf16(const float* __restrict__ src,
                           unsigned short* __restrict__ dst, int n) {
  int i = blockIdx.x * 256 + threadIdx.x;
  if (i < n) dst[i] = f2bf(src[i]);
}

// W: [K=EE, N=EE] row-major f32  ->  WT: [N, K] bf16
__global__ void k_transpose_bf16(const float* __restrict__ W,
                                 unsigned short* __restrict__ WT) {
  int i = blockIdx.x * 256 + threadIdx.x;
  if (i >= EE * EE) return;
  int n = i % EE, k = i / EE;                 // coalesced reads along n
  WT[n * EE + k] = f2bf(W[k * EE + n]);
}

// ---------------- wave-tile WMMA GEMM ----------------
// C[M=4096, N=1024] = A(bf16 row-major, K=1024) @ BT^T (+ bias)
// mode 0: bf16 out, [B,H,S,HD] layout     (Q, K)
// mode 1: bf16 out, [B,H,HD,S] layout     (V transposed)
// mode 2: f32  out, row-major [M,N]       (final projection)
__global__ __launch_bounds__(256) void k_gemm(
    const unsigned short* __restrict__ A,
    const unsigned short* __restrict__ BT,
    const float* __restrict__ bias,
    unsigned short* __restrict__ obf,
    float* __restrict__ of32,
    int mode) {
  const int lane = threadIdx.x & 31;
  const int wv   = threadIdx.x >> 5;          // 8 waves
  const int wm   = wv & 1, wn = wv >> 1;      // 2 (M) x 4 (N)
  const int row0 = blockIdx.x * 32 + wm * 16;
  const int col0 = blockIdx.y * 256 + wn * 64;
  const int l16  = lane & 15;
  const int hi   = lane >> 4;                 // 0/1 half of wave

  v8f acc[4];
#pragma unroll
  for (int t = 0; t < 4; ++t) acc[t] = vzero();

  const unsigned short* ap_row = A + (row0 + l16) * EE;

  for (int k0 = 0; k0 < EE; k0 += 32) {
    // A fragment: M=lane%16, K contiguous (0-15) / (16-31)
    const unsigned short* ap = ap_row + k0;
    v16bf af = make_frag(*(const u16x8*)(ap + (hi ? 16 : 0)),
                         *(const u16x8*)(ap + (hi ? 24 : 8)));
#pragma unroll
    for (int t = 0; t < 4; ++t) {
      int ncol = col0 + t * 16 + l16;
      const unsigned short* bp = BT + ncol * EE + k0 + (hi ? 16 : 0);
      v16bf bf = make_frag(*(const u16x8*)bp, *(const u16x8*)(bp + 8));
      acc[t] = bwmma(af, bf, acc[t]);
    }
  }

#pragma unroll
  for (int t = 0; t < 4; ++t) {
    int ncol = col0 + t * 16 + l16;
    float bv = bias ? bias[ncol] : 0.0f;
#pragma unroll
    for (int r = 0; r < 8; ++r) {
      int mrow = row0 + r + hi * 8;           // C row = vgpr + 8*half
      float v = acc[t][r] + bv;
      if (mode == 2) {
        of32[mrow * EE + ncol] = v;
      } else {
        int b = mrow >> 11, s = mrow & (SS - 1);
        int h = ncol >> 6,  hd = ncol & (HD - 1);
        int idx = (mode == 0) ? (((b * HH + h) * SS + s) * HD + hd)
                              : (((b * HH + h) * HD + hd) * SS + s);
        obf[idx] = f2bf(v);
      }
    }
  }
}

// ---------------- flash attention with relative bias ----------------
// one wave per 16-row q tile; kv chunks of 32; causal; rel bias via band GEMM
__global__ __launch_bounds__(128) void k_flash_attn(
    const unsigned short* __restrict__ Qb,   // [B*H, S, HD]
    const unsigned short* __restrict__ Kb,   // [B*H, S, HD]
    const unsigned short* __restrict__ Vt,   // [B*H, HD, S]
    const unsigned short* __restrict__ Rh,   // [P, HD]
    unsigned short* __restrict__ Ob)         // [B*S, E] row-major
{
  __shared__ unsigned short Plds[4][16][32]; // per-wave P tile (bf16)
  const int lane = threadIdx.x & 31;
  const int wv   = threadIdx.x >> 5;
  const int l16  = lane & 15;
  const int hi   = lane >> 4;
  const int bh   = blockIdx.y;
  const int b    = bh / HH, h = bh % HH;
  const int i0   = (blockIdx.x * 4 + wv) * 16;

  const unsigned short* Q = Qb + bh * (SS * HD);
  const unsigned short* K = Kb + bh * (SS * HD);
  const unsigned short* V = Vt + bh * (HD * SS);

  // Q A-fragments (hd 0..31 and 32..63)
  const unsigned short* qp = Q + (i0 + l16) * HD;
  v16bf qf0 = make_frag(*(const u16x8*)(qp + (hi ? 16 : 0)),
                        *(const u16x8*)(qp + (hi ? 24 : 8)));
  v16bf qf1 = make_frag(*(const u16x8*)(qp + 32 + (hi ? 16 : 0)),
                        *(const u16x8*)(qp + 32 + (hi ? 24 : 8)));

  v8f Oa[4];
#pragma unroll
  for (int t = 0; t < 4; ++t) Oa[t] = vzero();
  float mrow[8], lrow[8];
#pragma unroll
  for (int r = 0; r < 8; ++r) { mrow[r] = -3.0e38f; lrow[r] = 0.0f; }

  const float sc2 = 0.125f * 1.4426950408889634f;  // hd^-0.5 * log2(e)

  for (int j0 = 0; j0 <= i0 + 15; j0 += 32) {
    // ---- content scores: 16x32 tile = 2 N-tiles x 2 K-steps ----
    v8f Sf[2];
#pragma unroll
    for (int c = 0; c < 2; ++c) {
      Sf[c] = vzero();
      int j = j0 + c * 16 + l16;
      int jc = j < SS ? j : SS - 1;            // masked anyway if clamped
      const unsigned short* kp = K + jc * HD + (hi ? 16 : 0);
      v16bf bf0 = make_frag(*(const u16x8*)kp, *(const u16x8*)(kp + 8));
      Sf[c] = bwmma(qf0, bf0, Sf[c]);
      v16bf bf1 = make_frag(*(const u16x8*)(kp + 32), *(const u16x8*)(kp + 40));
      Sf[c] = bwmma(qf1, bf1, Sf[c]);
    }

    // ---- rel band: G = Q @ R_band^T, 47 used cols over 3 N-tiles ----
    v8f Gf[3];
    int base = (PP - 1) + j0 - i0 - 15;
#pragma unroll
    for (int c = 0; c < 3; ++c) {
      Gf[c] = vzero();
      int rr = base + c * 16 + l16;
      rr = rr < 0 ? 0 : (rr > PP - 1 ? PP - 1 : rr);
      const unsigned short* rp = Rh + rr * HD + (hi ? 16 : 0);
      v16bf bf0 = make_frag(*(const u16x8*)rp, *(const u16x8*)(rp + 8));
      Gf[c] = bwmma(qf0, bf0, Gf[c]);
      v16bf bf1 = make_frag(*(const u16x8*)(rp + 32), *(const u16x8*)(rp + 40));
      Gf[c] = bwmma(qf1, bf1, Gf[c]);
    }

    // ---- logits + causal mask + online softmax ----
    float pv[2][8];
#pragma unroll
    for (int r = 0; r < 8; ++r) {
      int a = r + hi * 8;
      float cm = -3.0e38f;
#pragma unroll
      for (int c = 0; c < 2; ++c) {
        int bcol = c * 16 + l16;
        int d = bcol - a + 15;                 // 0..46
        int idx = (d & 15) | (hi << 4);        // same row-half, col d%16
        float g0 = __shfl(Gf[0][r], idx, 32);
        float g1 = __shfl(Gf[1][r], idx, 32);
        float g2 = __shfl(Gf[2][r], idx, 32);
        int f = d >> 4;
        float g = (f == 0) ? g0 : ((f == 1) ? g1 : g2);
        float lv = (Sf[c][r] + g) * sc2;
        if (j0 + bcol > i0 + a) lv = -3.0e38f; // causal
        pv[c][r] = lv;
        cm = fmaxf(cm, lv);
      }
#pragma unroll
      for (int o = 1; o < 16; o <<= 1) cm = fmaxf(cm, __shfl_xor(cm, o, 32));
      float mnew  = fmaxf(mrow[r], cm);
      float alpha = exp2f(mrow[r] - mnew);
      mrow[r] = mnew;
      float ps = 0.0f;
#pragma unroll
      for (int c = 0; c < 2; ++c) {
        float p = exp2f(pv[c][r] - mnew);
        pv[c][r] = p;
        ps += p;
      }
#pragma unroll
      for (int o = 1; o < 16; o <<= 1) ps += __shfl_xor(ps, o, 32);
      lrow[r] = lrow[r] * alpha + ps;
#pragma unroll
      for (int t = 0; t < 4; ++t) Oa[t][r] *= alpha;
      // stage P (C-layout) into LDS for transpose to A-layout
      Plds[wv][a][l16]      = f2bf(pv[0][r]);
      Plds[wv][a][16 + l16] = f2bf(pv[1][r]);
    }
    asm volatile("s_wait_dscnt 0" ::: "memory");
    __builtin_amdgcn_wave_barrier();

    // P as 16x32 A-fragment
    const unsigned short* pp = &Plds[wv][l16][0];
    v16bf pf = make_frag(*(const u16x8*)(pp + (hi ? 16 : 0)),
                         *(const u16x8*)(pp + (hi ? 24 : 8)));
    __builtin_amdgcn_wave_barrier();

    // ---- O += P @ V  (V stored transposed: [HD, S]) ----
#pragma unroll
    for (int t = 0; t < 4; ++t) {
      int hdc = t * 16 + l16;
      int jb = j0 + (hi ? 16 : 0);
      if (jb > SS - 16) jb = SS - 16;          // OOB rows hit P==0
      const unsigned short* vp = V + hdc * SS + jb;
      v16bf bf = make_frag(*(const u16x8*)vp, *(const u16x8*)(vp + 8));
      Oa[t] = bwmma(pf, bf, Oa[t]);
    }
    asm volatile("s_wait_dscnt 0" ::: "memory");
    __builtin_amdgcn_wave_barrier();
  }

  // epilogue: O /= l, write bf16 row-major [B*S, E]
#pragma unroll
  for (int t = 0; t < 4; ++t) {
    int col = h * HD + t * 16 + l16;
#pragma unroll
    for (int r = 0; r < 8; ++r) {
      int srow = i0 + r + hi * 8;
      float v = Oa[t][r] / lrow[r];
      Ob[(b * SS + srow) * EE + col] = f2bf(v);
    }
  }
}

// ---------------- host launch ----------------

extern "C" void kernel_launch(void* const* d_in, const int* in_sizes, int n_in,
                              void* d_out, int out_size, void* d_ws, size_t ws_size,
                              hipStream_t stream) {
  const float* x   = (const float*)d_in[0];
  const float* Wq  = (const float*)d_in[1];
  const float* bq  = (const float*)d_in[2];
  const float* Wk  = (const float*)d_in[3];
  const float* bk  = (const float*)d_in[4];
  const float* Wv  = (const float*)d_in[5];
  const float* bv  = (const float*)d_in[6];
  const float* Wo  = (const float*)d_in[7];
  const float* bo  = (const float*)d_in[8];
  const float* rel = (const float*)d_in[9];

  char* ws = (char*)d_ws;
  unsigned short* Xh  = (unsigned short*)(ws);                    // 8 MiB
  unsigned short* WqT = (unsigned short*)(ws + (8u << 20));       // 2 MiB each
  unsigned short* WkT = (unsigned short*)(ws + (10u << 20));
  unsigned short* WvT = (unsigned short*)(ws + (12u << 20));
  unsigned short* WoT = (unsigned short*)(ws + (14u << 20));
  unsigned short* Rh  = (unsigned short*)(ws + (16u << 20));      // 256 KiB
  unsigned short* QbT = (unsigned short*)(ws + (17u << 20));      // 8 MiB
  unsigned short* KbT = (unsigned short*)(ws + (25u << 20));      // 8 MiB
  unsigned short* VtT = (unsigned short*)(ws + (33u << 20));      // 8 MiB
  unsigned short* ObT = (unsigned short*)(ws + (41u << 20));      // 8 MiB

  // prep: convert / transpose to bf16
  k_cvt_bf16<<<(MM * EE + 255) / 256, 256, 0, stream>>>(x, Xh, MM * EE);
  k_transpose_bf16<<<(EE * EE + 255) / 256, 256, 0, stream>>>(Wq, WqT);
  k_transpose_bf16<<<(EE * EE + 255) / 256, 256, 0, stream>>>(Wk, WkT);
  k_transpose_bf16<<<(EE * EE + 255) / 256, 256, 0, stream>>>(Wv, WvT);
  k_transpose_bf16<<<(EE * EE + 255) / 256, 256, 0, stream>>>(Wo, WoT);
  k_cvt_bf16<<<(PP * HD + 255) / 256, 256, 0, stream>>>(rel, Rh, PP * HD);

  // QKV projections (WMMA)
  dim3 gg(MM / 32, EE / 256);
  k_gemm<<<gg, 256, 0, stream>>>(Xh, WqT, bq, QbT, nullptr, 0);
  k_gemm<<<gg, 256, 0, stream>>>(Xh, WkT, bk, KbT, nullptr, 0);
  k_gemm<<<gg, 256, 0, stream>>>(Xh, WvT, bv, VtT, nullptr, 1);

  // flash attention with relative bias (WMMA)
  dim3 ga(SS / 64, BB * HH);
  k_flash_attn<<<ga, 128, 0, stream>>>(QbT, KbT, VtT, Rh, ObT);

  // output projection -> f32 d_out (WMMA)
  k_gemm<<<gg, 256, 0, stream>>>(ObT, WoT, bo, nullptr, (float*)d_out, 2);
}
